// MultiHeadCrossAttentionWithBias_60206851555965
// MI455X (gfx1250) — compile-verified, matrix operand order checked by reading, not run on hardware
//
#include <hip/hip_runtime.h>
#include <hip/hip_bf16.h>

// ---------------------------------------------------------------------------
// MultiHeadCrossAttentionWithBias for MI455X (gfx1250, wave32, WMMA bf16)
// B=4, LU=LE=1024, D=512, H=8, HD=64
// ---------------------------------------------------------------------------

#define L_SEQ 1024
#define DMODEL 512
#define NHEAD 8
#define HDIM 64
#define BATCH 4

typedef __bf16 bf16_t;
typedef __attribute__((ext_vector_type(16))) __bf16        v16bf;
typedef __attribute__((ext_vector_type(8)))  float         v8f;
typedef __attribute__((ext_vector_type(4)))  unsigned int  u32x4;
typedef __attribute__((ext_vector_type(4)))  float         f32x4;

union AFrag { v16bf v; u32x4 q[2]; };

__device__ __forceinline__ v8f wmma_bf16(v16bf a, v16bf b, v8f c) {
  return __builtin_amdgcn_wmma_f32_16x16x32_bf16(
      /*neg_a=*/false, a, /*neg_b=*/false, b,
      /*c_mod=*/(short)0, c, /*reuse_a=*/false, /*reuse_b=*/false);
}

// --- gfx1250 async copy global->LDS (ASYNCcnt-tracked, §15.18.3 op 98) -----
__device__ __forceinline__ unsigned lds_off32(const void* p) {
  return (unsigned)(unsigned long long)(
      __attribute__((address_space(3))) const void*)p;
}
__device__ __forceinline__ void async_copy_b128(void* lds_dst,
                                                const void* gsrc) {
  unsigned l = lds_off32(lds_dst);
  asm volatile("global_load_async_to_lds_b128 %0, %1, off"
               :
               : "v"(l), "v"(gsrc)
               : "memory");
}
#if __has_builtin(__builtin_amdgcn_s_wait_asynccnt)
#define WAIT_ASYNCCNT(n) __builtin_amdgcn_s_wait_asynccnt(n)
#else
#define WAIT_ASYNCCNT(n) asm volatile("s_wait_asynccnt %0" ::"i"(n) : "memory")
#endif

// ---------------------------------------------------------------------------
// Kernel 1: transpose+convert the 4 weight matrices to bf16 [N=512][K=512].
// wq/wk/wv flat: [k = d][n = h*64+hd]; wo flat: [k = h*64+hd][n = e].
// ---------------------------------------------------------------------------
__global__ __launch_bounds__(256)
void prep_weights(const float* __restrict__ wq, const float* __restrict__ wk,
                  const float* __restrict__ wv, const float* __restrict__ wo,
                  bf16_t* __restrict__ wT) {
  int idx = blockIdx.x * 256 + threadIdx.x;      // 4 * 512 * 512 total
  int w   = idx >> 18;
  int rem = idx & ((1 << 18) - 1);
  int n = rem >> 9, k = rem & 511;
  const float* src = (w == 0) ? wq : (w == 1) ? wk : (w == 2) ? wv : wo;
  wT[idx] = (bf16_t)src[k * 512 + n];            // wT[w][n][k]
}

// ---------------------------------------------------------------------------
// Kernel 2: projection GEMM  out = X[4096x512] * W[512x512] (+bias)
// Double-buffered LDS; B tile staged with async-to-LDS, A tile through VGPRs
// (needs f32->bf16 conversion). Copy of tile i+1 overlaps WMMAs of tile i.
// mode 0: q -> scale*(x+b), layout [B,H,L,HD]
// mode 1: k ->       (x+b), layout [B,H,L,HD]
// mode 2: v ->       (x+b), layout [B,H,HD,L]  (transposed for PV B-frags)
// ---------------------------------------------------------------------------
__global__ __launch_bounds__(256)
void proj_gemm(const float* __restrict__ X, const bf16_t* __restrict__ Wt,
               const float* __restrict__ bias, bf16_t* __restrict__ out,
               int mode) {
  __shared__ __align__(16) bf16_t Alds[2][128 * 40];
  __shared__ __align__(16) bf16_t Blds[2][128 * 40];

  const int tid  = threadIdx.x;
  const int lane = tid & 31;
  const int wave = tid >> 5;
  const int wm = wave & 1, wn = wave >> 1;   // 2 x 4 wave grid over 128x128
  const int m0 = blockIdx.x * 128;
  const int n0 = blockIdx.y * 128;
  const int lo = lane & 15;
  const int krun = (lane >> 4) * 8;          // A/B per-lane K run start
  const int row = tid >> 1, seg = tid & 1;   // 128 rows x 2 16-wide segments

  v8f acc[4][2];
#pragma unroll
  for (int i = 0; i < 4; i++)
#pragma unroll
    for (int j = 0; j < 2; j++)
#pragma unroll
      for (int e = 0; e < 8; e++) acc[i][j][e] = 0.f;

  auto stageA = [&](int kk, int buf) {
    const f32x4* asrc =
        (const f32x4*)(X + (size_t)(m0 + row) * DMODEL + kk + seg * 16);
    f32x4 a0 = asrc[0], a1 = asrc[1], a2 = asrc[2], a3 = asrc[3];
    bf16_t* dst = &Alds[buf][row * 40 + seg * 16];
#pragma unroll
    for (int e = 0; e < 4; e++) {
      dst[e]      = (bf16_t)a0[e];
      dst[4 + e]  = (bf16_t)a1[e];
      dst[8 + e]  = (bf16_t)a2[e];
      dst[12 + e] = (bf16_t)a3[e];
    }
  };
  auto stageB = [&](int kk, int buf) {
    const bf16_t* s = Wt + (size_t)(n0 + row) * DMODEL + kk + seg * 16;
    async_copy_b128(&Blds[buf][row * 40 + seg * 16], s);
    async_copy_b128(&Blds[buf][row * 40 + seg * 16 + 8], s + 8);
  };

  stageA(0, 0);
  stageB(0, 0);

  const int NIT = DMODEL / 32;  // 16
  for (int i = 0; i < NIT; i++) {
    const int cur = i & 1;
    __syncthreads();  // all waves done reading buf[cur^1] (iteration i-1)
    if (i + 1 < NIT) {
      stageA((i + 1) * 32, cur ^ 1);
      stageB((i + 1) * 32, cur ^ 1);
      WAIT_ASYNCCNT(2);  // tile i's async copies retired (in-order)
    } else {
      WAIT_ASYNCCNT(0);
    }
    __syncthreads();  // tile i fully visible in LDS

    AFrag a[4], b[2];
#pragma unroll
    for (int mt = 0; mt < 4; mt++) {
      int r = wm * 64 + mt * 16 + lo;
      a[mt].q[0] = *(const u32x4*)&Alds[cur][r * 40 + krun];
      a[mt].q[1] = *(const u32x4*)&Alds[cur][r * 40 + krun + 16];
    }
#pragma unroll
    for (int nt = 0; nt < 2; nt++) {
      int r = wn * 32 + nt * 16 + lo;
      b[nt].q[0] = *(const u32x4*)&Blds[cur][r * 40 + krun];
      b[nt].q[1] = *(const u32x4*)&Blds[cur][r * 40 + krun + 16];
    }
#pragma unroll
    for (int mt = 0; mt < 4; mt++)
#pragma unroll
      for (int nt = 0; nt < 2; nt++)
        acc[mt][nt] = wmma_bf16(a[mt].v, b[nt].v, acc[mt][nt]);
  }

  // ---- epilogue: bias, (scale), scatter to attention layout ----
#pragma unroll
  for (int nt = 0; nt < 2; nt++) {
    int n = n0 + wn * 32 + nt * 16 + lo;
    float bv = bias[n];
    int h = n >> 6, hd = n & 63;
#pragma unroll
    for (int mt = 0; mt < 4; mt++) {
#pragma unroll
      for (int j = 0; j < 8; j++) {
        int m    = m0 + wm * 64 + mt * 16 + j + ((lane >> 4) << 3);
        int bidx = m >> 10;
        int tok  = m & 1023;
        float v  = acc[mt][nt][j] + bv;
        if (mode == 0) v *= 0.125f;  // 1/sqrt(HD)
        size_t addr;
        if (mode == 2)
          addr = (((size_t)(bidx * NHEAD + h)) * HDIM + hd) * L_SEQ + tok;
        else
          addr = (((size_t)(bidx * NHEAD + h)) * L_SEQ + tok) * HDIM + hd;
        out[addr] = (bf16_t)v;
      }
    }
  }
}

// ---------------------------------------------------------------------------
// Kernel 3: flash cross-attention with additive bias + boolean mask.
// Block = 4 waves = 64 query rows of one (b,h); each wave owns 16 rows.
// K/V tiles for each 32-key chunk are staged into LDS once per block with
// gfx1250 async-to-LDS copies, double-buffered (copy i+1 overlaps compute i).
// Q,K: [B,H,L,HD] bf16; Vt: [B,H,HD,L] bf16; ctx out: [B,L,H*HD] bf16.
// ---------------------------------------------------------------------------
__global__ __launch_bounds__(128)
void attn(const bf16_t* __restrict__ Q, const bf16_t* __restrict__ K,
          const bf16_t* __restrict__ Vt, const float* __restrict__ logit_bpp,
          const unsigned char* __restrict__ mask, const float* __restrict__ pw,
          const float* __restrict__ pb, bf16_t* __restrict__ ctx,
          int transpose_bias) {
  __shared__ __align__(16) bf16_t Klds[2][32 * 72];  // 32 keys x 64 hd
  __shared__ __align__(16) bf16_t Vlds[2][64 * 40];  // 64 hd  x 32 keys
  __shared__ __align__(16) bf16_t Plds[4 * 16 * 40]; // wave-private P bounce

  const int tid  = threadIdx.x;
  const int lane = tid & 31;
  const int wave = tid >> 5;
  const int bh = blockIdx.y;            // 0..31
  const int b  = bh >> 3, h = bh & 7;
  const int q0 = blockIdx.x * 64 + wave * 16;
  const int lo = lane & 15;
  const int hi = lane >> 4;
  const int krun = hi * 8;

  const float bw = pw[0], bb = pb[0];

  const bf16_t* Qbase = Q + (size_t)bh * L_SEQ * HDIM;
  const bf16_t* Kbase = K + (size_t)bh * L_SEQ * HDIM;
  const bf16_t* Vbase = Vt + (size_t)bh * HDIM * L_SEQ;
  const unsigned char* Mbase = mask + (size_t)b * L_SEQ * L_SEQ;

  // per-thread async segments: K tile 32 rows x 4 segs, V tile 64 rows x 2
  const int krow = tid >> 2, kpart = tid & 3;
  const int vrow = tid >> 1, vpart = tid & 1;

  // Q fragments: 16 rows x 64 (K-dim) = 2 A-fragments, loaded once
  AFrag qf[2];
#pragma unroll
  for (int kk = 0; kk < 2; kk++) {
    const bf16_t* p = Qbase + (size_t)(q0 + lo) * HDIM + kk * 32 + krun;
    qf[kk].q[0] = *(const u32x4*)p;
    qf[kk].q[1] = *(const u32x4*)(p + 16);
  }

  v8f o[4];
#pragma unroll
  for (int i = 0; i < 4; i++)
#pragma unroll
    for (int e = 0; e < 8; e++) o[i][e] = 0.f;
  float rmax[8], rsum[8];
#pragma unroll
  for (int j = 0; j < 8; j++) { rmax[j] = -3.0e38f; rsum[j] = 0.f; }

  bf16_t* pl = &Plds[wave * 16 * 40];

  // prologue: stage chunk 0 into buffer 0
  async_copy_b128(&Klds[0][krow * 72 + kpart * 16],
                  Kbase + (size_t)krow * HDIM + kpart * 16);
  async_copy_b128(&Vlds[0][vrow * 40 + vpart * 16],
                  Vbase + (size_t)vrow * L_SEQ + vpart * 16);

  const int NCH = L_SEQ / 32;
  for (int ic = 0; ic < NCH; ic++) {
    const int kc  = ic * 32;
    const int cur = ic & 1;
    __syncthreads();  // all waves done reading buf[cur^1] (iteration ic-1)
    if (ic + 1 < NCH) {
      const int nkc = kc + 32;
      async_copy_b128(&Klds[cur ^ 1][krow * 72 + kpart * 16],
                      Kbase + (size_t)(nkc + krow) * HDIM + kpart * 16);
      async_copy_b128(&Vlds[cur ^ 1][vrow * 40 + vpart * 16],
                      Vbase + (size_t)vrow * L_SEQ + nkc + vpart * 16);
      WAIT_ASYNCCNT(2);  // chunk ic's 2 copies retired (in-order completion)
    } else {
      WAIT_ASYNCCNT(0);
    }
    __syncthreads();  // every wave's chunk-ic copies visible in LDS

    // ---- S = Q*K^T : two 16x16 tiles, K-dim 64, K-frags from LDS ----
    v8f s[2];
#pragma unroll
    for (int c = 0; c < 2; c++) {
#pragma unroll
      for (int e = 0; e < 8; e++) s[c][e] = 0.f;
#pragma unroll
      for (int kk = 0; kk < 2; kk++) {
        AFrag kf;
        const bf16_t* p = &Klds[cur][(c * 16 + lo) * 72 + kk * 32 + krun];
        kf.q[0] = *(const u32x4*)p;
        kf.q[1] = *(const u32x4*)(p + 16);
        s[c] = wmma_bf16(qf[kk].v, kf.v, s[c]);
      }
    }
    // ---- bias + mask (C layout: row = j + 8*hi, col = lo) ----
#pragma unroll
    for (int c = 0; c < 2; c++) {
      int kg = kc + c * 16 + lo;
#pragma unroll
      for (int j = 0; j < 8; j++) {
        int qg = q0 + j + hi * 8;
        float bias = transpose_bias ? logit_bpp[(size_t)kg * L_SEQ + qg]
                                    : logit_bpp[(size_t)qg * L_SEQ + kg];
        float v = s[c][j] + bw * bias + bb;
        if (!Mbase[(size_t)qg * L_SEQ + kg]) v = -3.0e38f;
        s[c][j] = v;
      }
    }
    // ---- online softmax: row reduce across the 16-lane N group ----
    float mnew[8];
#pragma unroll
    for (int j = 0; j < 8; j++) {
      float mx = fmaxf(s[0][j], s[1][j]);
#pragma unroll
      for (int d = 1; d < 16; d <<= 1) mx = fmaxf(mx, __shfl_xor(mx, d, 32));
      mnew[j] = fmaxf(rmax[j], mx);
    }
#pragma unroll
    for (int j = 0; j < 8; j++) {
      float alpha = __expf(rmax[j] - mnew[j]);
      rmax[j] = mnew[j];
      float p0 = __expf(s[0][j] - mnew[j]);
      float p1 = __expf(s[1][j] - mnew[j]);
      s[0][j] = p0; s[1][j] = p1;
      float ps = p0 + p1;
#pragma unroll
      for (int d = 1; d < 16; d <<= 1) ps += __shfl_xor(ps, d, 32);
      rsum[j] = rsum[j] * alpha + ps;
#pragma unroll
      for (int t = 0; t < 4; t++) o[t][j] *= alpha;
    }
    // ---- relayout P: C-layout regs -> wave-private LDS -> A-fragment.
    // Same-wave LDS ops are in-order; only a compiler fence is needed. ----
#pragma unroll
    for (int c = 0; c < 2; c++)
#pragma unroll
      for (int j = 0; j < 8; j++)
        pl[(j + hi * 8) * 40 + c * 16 + lo] = (bf16_t)s[c][j];
    asm volatile("" ::: "memory");
    AFrag pf;
    {
      const bf16_t* p = pl + lo * 40 + krun;
      pf.q[0] = *(const u32x4*)p;
      pf.q[1] = *(const u32x4*)(p + 16);
    }
    // ---- O += P * V : V-frags from LDS (rows = head-dims) ----
#pragma unroll
    for (int nt = 0; nt < 4; nt++) {
      AFrag vf;
      const bf16_t* p = &Vlds[cur][(nt * 16 + lo) * 40 + krun];
      vf.q[0] = *(const u32x4*)p;
      vf.q[1] = *(const u32x4*)(p + 16);
      o[nt] = wmma_bf16(pf.v, vf.v, o[nt]);
    }
  }

  // ---- finalize: divide by rowsum, store ctx[b][tok][h*64+hd] ----
  bf16_t* cb = ctx + (size_t)b * L_SEQ * DMODEL + (size_t)h * HDIM;
#pragma unroll
  for (int j = 0; j < 8; j++) {
    float inv = 1.0f / rsum[j];
    int tok = q0 + j + hi * 8;
#pragma unroll
    for (int nt = 0; nt < 4; nt++) {
      int hd = nt * 16 + lo;
      cb[(size_t)tok * DMODEL + hd] = (bf16_t)(o[nt][j] * inv);
    }
  }
}

// ---------------------------------------------------------------------------
// Kernel 4: output projection  out = ctx[4096x512](bf16) * woT + wo_b  (f32)
// Both tiles bf16 -> both staged with double-buffered async-to-LDS copies.
// ---------------------------------------------------------------------------
__global__ __launch_bounds__(256)
void out_gemm(const bf16_t* __restrict__ Xb, const bf16_t* __restrict__ Wt,
              const float* __restrict__ bias, float* __restrict__ out) {
  __shared__ __align__(16) bf16_t Alds[2][128 * 40];
  __shared__ __align__(16) bf16_t Blds[2][128 * 40];

  const int tid  = threadIdx.x;
  const int lane = tid & 31;
  const int wave = tid >> 5;
  const int wm = wave & 1, wn = wave >> 1;
  const int m0 = blockIdx.x * 128;
  const int n0 = blockIdx.y * 128;
  const int lo = lane & 15;
  const int krun = (lane >> 4) * 8;
  const int row = tid >> 1, seg = tid & 1;

  v8f acc[4][2];
#pragma unroll
  for (int i = 0; i < 4; i++)
#pragma unroll
    for (int j = 0; j < 2; j++)
#pragma unroll
      for (int e = 0; e < 8; e++) acc[i][j][e] = 0.f;

  auto stage = [&](int kk, int buf) {
    const bf16_t* a = Xb + (size_t)(m0 + row) * DMODEL + kk + seg * 16;
    const bf16_t* s = Wt + (size_t)(n0 + row) * DMODEL + kk + seg * 16;
    async_copy_b128(&Alds[buf][row * 40 + seg * 16], a);
    async_copy_b128(&Alds[buf][row * 40 + seg * 16 + 8], a + 8);
    async_copy_b128(&Blds[buf][row * 40 + seg * 16], s);
    async_copy_b128(&Blds[buf][row * 40 + seg * 16 + 8], s + 8);
  };

  stage(0, 0);

  const int NIT = DMODEL / 32;  // 16
  for (int i = 0; i < NIT; i++) {
    const int cur = i & 1;
    __syncthreads();  // all waves done reading buf[cur^1]
    if (i + 1 < NIT) {
      stage((i + 1) * 32, cur ^ 1);
      WAIT_ASYNCCNT(4);  // tile i's 4 async copies retired (in-order)
    } else {
      WAIT_ASYNCCNT(0);
    }
    __syncthreads();  // tile i fully visible in LDS

    AFrag a[4], b[2];
#pragma unroll
    for (int mt = 0; mt < 4; mt++) {
      int r = wm * 64 + mt * 16 + lo;
      a[mt].q[0] = *(const u32x4*)&Alds[cur][r * 40 + krun];
      a[mt].q[1] = *(const u32x4*)&Alds[cur][r * 40 + krun + 16];
    }
#pragma unroll
    for (int nt = 0; nt < 2; nt++) {
      int r = wn * 32 + nt * 16 + lo;
      b[nt].q[0] = *(const u32x4*)&Blds[cur][r * 40 + krun];
      b[nt].q[1] = *(const u32x4*)&Blds[cur][r * 40 + krun + 16];
    }
#pragma unroll
    for (int mt = 0; mt < 4; mt++)
#pragma unroll
      for (int nt = 0; nt < 2; nt++)
        acc[mt][nt] = wmma_bf16(a[mt].v, b[nt].v, acc[mt][nt]);
  }

#pragma unroll
  for (int nt = 0; nt < 2; nt++) {
    int n = n0 + wn * 32 + nt * 16 + lo;
    float bv = bias[n];
#pragma unroll
    for (int mt = 0; mt < 4; mt++) {
#pragma unroll
      for (int j = 0; j < 8; j++) {
        int m = m0 + wm * 64 + mt * 16 + j + ((lane >> 4) << 3);
        out[(size_t)m * DMODEL + n] = acc[mt][nt][j] + bv;
      }
    }
  }
}

// ---------------------------------------------------------------------------
// Host-side orchestration
// ---------------------------------------------------------------------------
extern "C" void kernel_launch(void* const* d_in, const int* in_sizes, int n_in,
                              void* d_out, int out_size, void* d_ws,
                              size_t ws_size, hipStream_t stream) {
  (void)in_sizes; (void)n_in; (void)out_size; (void)ws_size;

  const float* u_enc     = (const float*)d_in[0];
  const float* e_enc     = (const float*)d_in[1];
  const float* logit_bpp = (const float*)d_in[2];
  const unsigned char* ue_mask = (const unsigned char*)d_in[3];
  const unsigned char* eu_mask = (const unsigned char*)d_in[4];
  const float* wq_k = (const float*)d_in[5];
  const float* wq_b = (const float*)d_in[6];
  const float* wk_k = (const float*)d_in[7];
  const float* wk_b = (const float*)d_in[8];
  const float* wv_k = (const float*)d_in[9];
  const float* wv_b = (const float*)d_in[10];
  const float* wo_k = (const float*)d_in[11];
  const float* wo_b = (const float*)d_in[12];
  const float* bpp_w = (const float*)d_in[13];
  const float* bpp_b = (const float*)d_in[14];

  // workspace layout (bf16 elements)
  const size_t QKV = (size_t)BATCH * NHEAD * L_SEQ * HDIM;  // 2,097,152
  const size_t WSZ = (size_t)DMODEL * DMODEL;               //   262,144
  bf16_t* ws   = (bf16_t*)d_ws;
  bf16_t* q_u  = ws + 0 * QKV;
  bf16_t* k_u  = ws + 1 * QKV;
  bf16_t* vT_u = ws + 2 * QKV;
  bf16_t* q_e  = ws + 3 * QKV;
  bf16_t* k_e  = ws + 4 * QKV;
  bf16_t* vT_e = ws + 5 * QKV;
  bf16_t* wT   = ws + 6 * QKV;      // wqT, wkT, wvT, woT back-to-back
  bf16_t* ctx_u = wT + 4 * WSZ;
  bf16_t* ctx_e = ctx_u + QKV;

  // 1) weights -> bf16, transposed to [N][K]
  prep_weights<<<4096, 256, 0, stream>>>(wq_k, wk_k, wv_k, wo_k, wT);

  // 2) QKV projections (6 GEMMs)
  dim3 pg(32, 4);
  proj_gemm<<<pg, 256, 0, stream>>>(u_enc, wT + 0 * WSZ, wq_b, q_u, 0);
  proj_gemm<<<pg, 256, 0, stream>>>(u_enc, wT + 1 * WSZ, wk_b, k_u, 1);
  proj_gemm<<<pg, 256, 0, stream>>>(u_enc, wT + 2 * WSZ, wv_b, vT_u, 2);
  proj_gemm<<<pg, 256, 0, stream>>>(e_enc, wT + 0 * WSZ, wq_b, q_e, 0);
  proj_gemm<<<pg, 256, 0, stream>>>(e_enc, wT + 1 * WSZ, wk_b, k_e, 1);
  proj_gemm<<<pg, 256, 0, stream>>>(e_enc, wT + 2 * WSZ, wv_b, vT_e, 2);

  // 3) flash cross-attention, both directions
  dim3 ag(16, 32);
  attn<<<ag, 128, 0, stream>>>(q_u, k_e, vT_e, logit_bpp, ue_mask, bpp_w,
                               bpp_b, ctx_u, 0);
  attn<<<ag, 128, 0, stream>>>(q_e, k_u, vT_u, logit_bpp, eu_mask, bpp_w,
                               bpp_b, ctx_e, 1);

  // 4) output projections straight into d_out (u_update, then e_update)
  float* out = (float*)d_out;
  out_gemm<<<pg, 256, 0, stream>>>(ctx_u, wT + 3 * WSZ, wo_b, out);
  out_gemm<<<pg, 256, 0, stream>>>(ctx_e, wT + 3 * WSZ, wo_b,
                                   out + (size_t)BATCH * L_SEQ * DMODEL);
}
